// SA_Layer_Single_Head_910533067685
// MI455X (gfx1250) — matrix operand order repaired and checked by reading, not compiled, not run on hardware
//
#include <hip/hip_runtime.h>

// ---------------------------------------------------------------------------
// SA offset-attention layer for MI455X (gfx1250), wave32 + WMMA f16->f32.
// Flash-style: the (B,N,N) attention matrix is never written to HBM.
// x^T tile staging into LDS is done by the Tensor Data Mover (TDM) with
// native LDS padding (stride 40 halves), falling back to manual staging if
// the tensor builtin is unavailable.
// ---------------------------------------------------------------------------

typedef _Float16 h8    __attribute__((ext_vector_type(8)));
typedef _Float16 v16h  __attribute__((ext_vector_type(16)));
typedef _Float16 h2    __attribute__((ext_vector_type(2)));
typedef float    v8f   __attribute__((ext_vector_type(8)));
typedef unsigned u32x4 __attribute__((ext_vector_type(4)));
typedef int      i32x4 __attribute__((ext_vector_type(4)));
typedef int      i32x8 __attribute__((ext_vector_type(8)));

constexpr int B_ = 16, N_ = 2048, C_ = 256, C4_ = 64;
#define XTS 40    // LDS stride (halves) for x^T tile: 80B -> 16B aligned, conflict-free
#define DTS 264   // LDS stride (halves) for d tile:   528B -> 16B aligned, conflict-free

#if defined(__gfx1250__) && __has_builtin(__builtin_amdgcn_tensor_load_to_lds) && \
    __has_builtin(__builtin_amdgcn_s_wait_tensorcnt)
#define USE_TDM 1
#else
#define USE_TDM 0
#endif

__device__ __forceinline__ v16h cat16(h8 a, h8 b) {
  return __builtin_shufflevector(a, b, 0,1,2,3,4,5,6,7,8,9,10,11,12,13,14,15);
}
__device__ __forceinline__ v8f wmma(v16h a, v16h b, v8f c) {
  return __builtin_amdgcn_wmma_f32_16x16x32_f16(false, a, false, b, (short)0, c,
                                                false, false);
}
// A-operand gather: lane holds row M=l&15; K elems {koff..koff+15} contiguous
__device__ __forceinline__ v16h loadA(const _Float16* p) {
  return cat16(*(const h8*)p, *(const h8*)(p + 8));
}

// ---------------------------------------------------------------------------
// Kernel 1: f32 -> f16 conversions. xh row-major (B,N,C); xhT transposed
// (B,C,N) copy feeding the TDM tile loads; weights row-major.
// ---------------------------------------------------------------------------
__global__ void __launch_bounds__(256) prep_kernel(
    const float* __restrict__ x, const float* __restrict__ Wq,
    const float* __restrict__ Wk, const float* __restrict__ Wt,
    _Float16* __restrict__ xh, _Float16* __restrict__ xhT,
    _Float16* __restrict__ Wqh, _Float16* __restrict__ Wkh,
    _Float16* __restrict__ Wth)
{
  long i = (long)blockIdx.x * 256 + threadIdx.x;
  if (i < (long)B_ * N_ * C_) {
    _Float16 v = (_Float16)x[i];
    xh[i] = v;
    int c = (int)(i & (C_ - 1));
    int n = (int)((i >> 8) & (N_ - 1));
    int b = (int)(i >> 19);
    xhT[((long)b * C_ + c) * N_ + n] = v;
  }
  if (i < C4_ * C_) { Wqh[i] = (_Float16)Wq[i]; Wkh[i] = (_Float16)Wk[i]; }
  if (i < C_ * C_)  { Wth[i] = (_Float16)Wt[i]; }
}

// ---------------------------------------------------------------------------
// Kernel 2: Qh = xh*Wq^T, Kh = xh*Wk^T  (f16 out, row-major (N,64) per batch).
// One 16-row tile per wave; K-loop over C in chunks of 32; 64 WMMAs/wave.
// ---------------------------------------------------------------------------
__global__ void __launch_bounds__(256) qk_kernel(
    const _Float16* __restrict__ xh, const _Float16* __restrict__ Wqh,
    const _Float16* __restrict__ Wkh,
    _Float16* __restrict__ Qh, _Float16* __restrict__ Kh)
{
  const int b = blockIdx.x >> 4, grp = blockIdx.x & 15;
  const int wave = threadIdx.x >> 5, lane = threadIdx.x & 31;
  const int l15 = lane & 15, hi = lane >> 4, koff = hi * 16;
  const int n0 = (grp * 8 + wave) * 16;
  const _Float16* xrow = xh + ((long)b * N_ + n0 + l15) * C_;
  v8f accQ[4] = {}, accK[4] = {};
  for (int kc = 0; kc < 8; ++kc) {
    v16h a = loadA(xrow + kc * 32 + koff);
    #pragma unroll
    for (int ot = 0; ot < 4; ++ot) {
      v16h bq = *(const v16h*)(Wqh + (ot * 16 + l15) * C_ + kc * 32 + hi * 16);
      accQ[ot] = wmma(a, bq, accQ[ot]);
      v16h bk = *(const v16h*)(Wkh + (ot * 16 + l15) * C_ + kc * 32 + hi * 16);
      accK[ot] = wmma(a, bk, accK[ot]);
    }
  }
  #pragma unroll
  for (int ot = 0; ot < 4; ++ot)
    #pragma unroll
    for (int v = 0; v < 8; ++v) {
      long o = ((long)b * N_ + n0 + v + hi * 8) * C4_ + ot * 16 + l15;
      Qh[o] = (_Float16)accQ[ot][v];
      Kh[o] = (_Float16)accK[ot][v];
    }
}

// ---------------------------------------------------------------------------
// Kernel 3: row softmax statistics. Two WMMA passes over all m-tiles:
// pass1 row max, pass2 sum of exp. Cross-lane reduce via shfl_xor (wave32).
// ---------------------------------------------------------------------------
__global__ void __launch_bounds__(256) stats_kernel(
    const _Float16* __restrict__ Qh, const _Float16* __restrict__ Kh,
    float* __restrict__ rowmax, float* __restrict__ rsinv)
{
  const int b = blockIdx.x >> 4, grp = blockIdx.x & 15;
  const int wave = threadIdx.x >> 5, lane = threadIdx.x & 31;
  const int l15 = lane & 15, hi = lane >> 4, koff = hi * 16;
  const int n0 = (grp * 8 + wave) * 16;
  const _Float16* qrow = Qh + ((long)b * N_ + n0 + l15) * C4_;
  v16h aq0 = loadA(qrow + koff);
  v16h aq1 = loadA(qrow + 32 + koff);
  const _Float16* kb = Kh + (long)b * N_ * C4_;

  float rmax[8];
  #pragma unroll
  for (int v = 0; v < 8; ++v) rmax[v] = -3.0e38f;
  for (int mt = 0; mt < N_ / 16; ++mt) {
    const _Float16* krow = kb + (mt * 16 + l15) * C4_ + hi * 16;
    v8f e = {};
    e = wmma(aq0, *(const v16h*)krow, e);
    e = wmma(aq1, *(const v16h*)(krow + 32), e);
    #pragma unroll
    for (int v = 0; v < 8; ++v) rmax[v] = fmaxf(rmax[v], e[v]);
  }
  for (int off = 1; off < 16; off <<= 1)
    #pragma unroll
    for (int v = 0; v < 8; ++v) rmax[v] = fmaxf(rmax[v], __shfl_xor(rmax[v], off, 32));

  float rs[8] = {};
  for (int mt = 0; mt < N_ / 16; ++mt) {
    const _Float16* krow = kb + (mt * 16 + l15) * C4_ + hi * 16;
    v8f e = {};
    e = wmma(aq0, *(const v16h*)krow, e);
    e = wmma(aq1, *(const v16h*)(krow + 32), e);
    #pragma unroll
    for (int v = 0; v < 8; ++v) rs[v] += __expf(e[v] - rmax[v]);
  }
  for (int off = 1; off < 16; off <<= 1)
    #pragma unroll
    for (int v = 0; v < 8; ++v) rs[v] += __shfl_xor(rs[v], off, 32);

  if (l15 == 0) {
    #pragma unroll
    for (int v = 0; v < 8; ++v) {
      long n = (long)b * N_ + n0 + v + hi * 8;
      rowmax[n] = rmax[v];
      rsinv[n]  = 1.0f / rs[v];
    }
  }
}

// ---------------------------------------------------------------------------
// Kernel 4 (fused): per wave, one m-tile of 16 output rows.
//   loop n in chunks of 32:  TDM stages x^T tile -> energy(WMMA) ->
//   exp/softmax -> LDS transpose -> x_r += attn^T * x (WMMA) + colsums.
//   then: scale by 1/(1e-6+colsum), d = x - x_r, t = d*Wt^T (WMMA),
//   BN + ReLU + residual, store out.
// ---------------------------------------------------------------------------
__global__ void __launch_bounds__(128) fused_kernel(
    const float* __restrict__ x, const _Float16* __restrict__ xh,
    const _Float16* __restrict__ xhT,
    const _Float16* __restrict__ Qh, const _Float16* __restrict__ Kh,
    const _Float16* __restrict__ Wth,
    const float* __restrict__ rowmax, const float* __restrict__ rsinv,
    const float* __restrict__ bt, const float* __restrict__ gamma,
    const float* __restrict__ beta, const float* __restrict__ bn_mean,
    const float* __restrict__ bn_var, float* __restrict__ out)
{
  __shared__ _Float16 sXT[C_ * XTS];      // x^T tile (c-major) for 32 n's
  __shared__ _Float16 sAT[4][16 * XTS];   // per-wave attn^T (m x 32n)
  __shared__ _Float16 sD[4][16 * DTS];    // per-wave d tile (m x C)
  __shared__ float sRmax[32];
  __shared__ float sRs[32];

  const int b = blockIdx.x >> 5, grp = blockIdx.x & 31;
  const int tid = threadIdx.x;
  const int wave = tid >> 5, lane = tid & 31;
  const int l15 = lane & 15, hi = lane >> 4, koff = hi * 16;
  const int m0 = (grp * 4 + wave) * 16;

  // energy B-operand: columns = this wave's m-tile (kept in regs all loop)
  const _Float16* kb = Kh + (long)b * N_ * C4_ + (m0 + l15) * C4_ + hi * 16;
  v16h kB0 = *(const v16h*)kb;
  v16h kB1 = *(const v16h*)(kb + 32);
  const _Float16* qb = Qh + (long)b * N_ * C4_;

  v8f xr[16] = {};
  float colacc = 0.f;

  for (int nc = 0; nc < N_; nc += 32) {
    __syncthreads();                      // previous sXT fully consumed
#if USE_TDM
    if (wave == 0) {
      // Tensor DMA: 2D tile (32 x 256 halves) from xhT, LDS-padded to XTS=40.
      unsigned long long ga =
          (unsigned long long)(const void*)(xhT + (long)b * C_ * N_ + nc);
      unsigned ldsOff = (unsigned)(unsigned long long)(const void*)&sXT[0];
      // D# group0: count=1 | lds_addr | global_addr(57b) | type=2
      u32x4 g0 = { 1u, ldsOff, (unsigned)ga,
                   (unsigned)((ga >> 32) & 0x1FFFFFFu) | (2u << 30) };
      // D# group1: data_size=2B, pad_enable, pad_interval=16dw, pad_amount=4dw,
      //            tensor_dim0=N, tensor_dim1=C, tile 32x256, stride0=N
      i32x8 g1 = { (int)((1u << 16) | (1u << 20) | (3u << 22) | (3u << 25)),
                   (int)((unsigned)N_ << 16), (int)((unsigned)C_ << 16),
                   (int)(32u << 16), (int)C_, (int)N_, 0, 0 };
      i32x4 z4 = { 0, 0, 0, 0 };
#if __clang_major__ >= 23
      i32x8 z8 = { 0, 0, 0, 0, 0, 0, 0, 0 };
      __builtin_amdgcn_tensor_load_to_lds(g0, g1, z4, z4, z8, 0);
#else
      __builtin_amdgcn_tensor_load_to_lds(g0, g1, z4, z4, 0);
#endif
      if (tid < 32) {
        sRmax[tid] = rowmax[(long)b * N_ + nc + tid];
        sRs[tid]   = rsinv [(long)b * N_ + nc + tid];
      }
      __builtin_amdgcn_s_wait_tensorcnt(0);
    }
    __syncthreads();
#else
    if (tid < 32) {
      sRmax[tid] = rowmax[(long)b * N_ + nc + tid];
      sRs[tid]   = rsinv [(long)b * N_ + nc + tid];
    }
    __syncthreads();
    for (int idx = tid; idx < 32 * C_; idx += 128) {
      int c = idx & (C_ - 1), nn = idx >> 8;
      sXT[c * XTS + nn] = xh[((long)b * N_ + nc + nn) * C_ + c];
    }
    __syncthreads();
#endif

    _Float16* atw = sAT[wave];
    #pragma unroll
    for (int sub = 0; sub < 2; ++sub) {
      // energy tile (16n x 16m), K = 64 over o
      const _Float16* qrow = qb + (long)(nc + sub * 16 + l15) * C4_ + koff;
      v8f e = {};
      e = wmma(loadA(qrow), kB0, e);
      e = wmma(loadA(qrow + 32), kB1, e);
      // softmax + write transposed (m-major) to per-wave LDS
      #pragma unroll
      for (int v = 0; v < 8; v += 2) {
        int nn = sub * 16 + hi * 8 + v;
        float a0 = __expf(e[v]     - sRmax[nn])     * sRs[nn];
        float a1 = __expf(e[v + 1] - sRmax[nn + 1]) * sRs[nn + 1];
        colacc += a0 + a1;
        h2 p; p[0] = (_Float16)a0; p[1] = (_Float16)a1;
        *(h2*)(atw + l15 * XTS + nn) = p;
      }
    }
    // x_r accumulation: A = attn^T (16m x 32n) from LDS, B = x^T tile
    v16h aA = cat16(*(const h8*)(atw + l15 * XTS + koff),
                    *(const h8*)(atw + l15 * XTS + koff + 8));
    #pragma unroll
    for (int ct = 0; ct < 16; ++ct) {
      const _Float16* xp = sXT + (ct * 16 + l15) * XTS + hi * 16;
      v16h bx = cat16(*(const h8*)xp, *(const h8*)(xp + 8));
      xr[ct] = wmma(aA, bx, xr[ct]);
    }
  }

  // column-sum double normalization: scale x_r row m by 1/(1e-6 + colsum[m])
  colacc += __shfl_xor(colacc, 16, 32);
  float scale = 1.0f / (1e-6f + colacc);    // held by lane pair (l15, l15+16)
  float scl[8];
  #pragma unroll
  for (int v = 0; v < 8; ++v) scl[v] = __shfl(scale, v + hi * 8, 32);

  // d = x - x_r  -> per-wave LDS, f16 row-major
  const float* xb = x + ((long)b * N_ + m0) * C_;
  _Float16* dw = sD[wave];
  #pragma unroll
  for (int ct = 0; ct < 16; ++ct) {
    #pragma unroll
    for (int v = 0; v < 8; ++v) {
      int row = v + hi * 8, c = ct * 16 + l15;
      float dv = xb[row * C_ + c] - xr[ct][v] * scl[v];
      dw[row * DTS + c] = (_Float16)dv;
    }
  }

  // t = d * Wt^T  (16 x 256, K = 256)
  v8f tacc[16] = {};
  for (int kc = 0; kc < 8; ++kc) {
    v16h aD = cat16(*(const h8*)(dw + l15 * DTS + kc * 32 + koff),
                    *(const h8*)(dw + l15 * DTS + kc * 32 + koff + 8));
    #pragma unroll
    for (int ot = 0; ot < 16; ++ot) {
      v16h bw = *(const v16h*)(Wth + (ot * 16 + l15) * C_ + kc * 32 + hi * 16);
      tacc[ot] = wmma(aD, bw, tacc[ot]);
    }
  }

  // epilogue: bias + BN + ReLU + residual
  float* ob = out + ((long)b * N_ + m0) * C_;
  #pragma unroll
  for (int ot = 0; ot < 16; ++ot) {
    int o = ot * 16 + l15;
    float btv = bt[o], g = gamma[o], be = beta[o], mu = bn_mean[o];
    float rstd = rsqrtf(bn_var[o] + 1e-5f);
    #pragma unroll
    for (int v = 0; v < 8; ++v) {
      int row = v + hi * 8;
      float tv = tacc[ot][v] + btv;
      float bn = g * (tv - mu) * rstd + be;
      ob[row * C_ + o] = xb[row * C_ + o] + fmaxf(bn, 0.f);
    }
  }
}

// ---------------------------------------------------------------------------
extern "C" void kernel_launch(void* const* d_in, const int* in_sizes, int n_in,
                              void* d_out, int out_size, void* d_ws, size_t ws_size,
                              hipStream_t stream) {
  (void)in_sizes; (void)n_in; (void)out_size; (void)ws_size;
  const float* x     = (const float*)d_in[0];
  const float* Wq    = (const float*)d_in[1];
  const float* Wk    = (const float*)d_in[2];
  // d_in[3] = Wv, d_in[4] = bv : dead in the reference forward pass
  const float* Wt    = (const float*)d_in[5];
  const float* bt    = (const float*)d_in[6];
  const float* gamma = (const float*)d_in[7];
  const float* beta  = (const float*)d_in[8];
  const float* bmean = (const float*)d_in[9];
  const float* bvar  = (const float*)d_in[10];
  float* out = (float*)d_out;

  char* p = (char*)d_ws;
  auto take = [&](size_t bytes) {
    char* r = p; p += (bytes + 255) & ~size_t(255); return r;
  };
  _Float16* xh   = (_Float16*)take((size_t)B_ * N_ * C_ * 2);   // 16 MB
  _Float16* xhT  = (_Float16*)take((size_t)B_ * N_ * C_ * 2);   // 16 MB
  _Float16* Wqh  = (_Float16*)take((size_t)C4_ * C_ * 2);
  _Float16* Wkh  = (_Float16*)take((size_t)C4_ * C_ * 2);
  _Float16* Wth  = (_Float16*)take((size_t)C_ * C_ * 2);
  _Float16* Qh   = (_Float16*)take((size_t)B_ * N_ * C4_ * 2);  // 4 MB
  _Float16* Kh   = (_Float16*)take((size_t)B_ * N_ * C4_ * 2);  // 4 MB
  float* rowmax  = (float*)take((size_t)B_ * N_ * 4);
  float* rsinv   = (float*)take((size_t)B_ * N_ * 4);

  long totalx = (long)B_ * N_ * C_;
  prep_kernel<<<(int)((totalx + 255) / 256), 256, 0, stream>>>(
      x, Wq, Wk, Wt, xh, xhT, Wqh, Wkh, Wth);
  qk_kernel<<<B_ * 16, 256, 0, stream>>>(xh, Wqh, Wkh, Qh, Kh);
  stats_kernel<<<B_ * 16, 256, 0, stream>>>(Qh, Kh, rowmax, rsinv);
  fused_kernel<<<B_ * 32, 128, 0, stream>>>(x, xh, xhT, Qh, Kh, Wth, rowmax,
                                            rsinv, bt, gamma, beta, bmean,
                                            bvar, out);
}
